// InvariantPointAttention_23416161698182
// MI455X (gfx1250) — matrix operand order repaired and checked by reading, not compiled, not run on hardware
//
#include <hip/hip_runtime.h>
#include <hip/hip_bf16.h>
#include <math.h>

typedef __attribute__((ext_vector_type(16))) __bf16 v16bf;
typedef __attribute__((ext_vector_type(8)))  float  v8f;
typedef int v4i __attribute__((vector_size(16)));

#define kN   512
#define kH   12
#define kCS  384
#define kCZ  128
#define kDQK 16
#define kDV  16
#define kPQK 4
#define kPV  8
#define kCAT 2112   /* H * (DV + PV*4 + CZ) = 12 * 176 */

__constant__ const float kSW  = 0.14433756729740643f;   /* sqrt(1/48)  */
__constant__ const float kPW  = 0.13608276348795434f;   /* sqrt(1/54)  */
__constant__ const float kZW  = 0.57735026918962576f;   /* sqrt(1/3)   */
__constant__ const float kEPS = 1e-8f;
__constant__ const float kNEG = -3.3895313892515355e38f;

/* ---------------- async global -> LDS staging (CDNA5 path) --------------- */

__device__ __forceinline__ void async_copy_f4(const float* gsrc, float* ldst) {
#if __has_builtin(__builtin_amdgcn_global_load_async_to_lds_b128)
  __builtin_amdgcn_global_load_async_to_lds_b128(
      (__attribute__((address_space(1))) v4i*)gsrc,
      (__attribute__((address_space(3))) v4i*)ldst, 0, 0);
#else
  *(float4*)ldst = *(const float4*)gsrc;
#endif
}

__device__ __forceinline__ void async_join() {
#if __has_builtin(__builtin_amdgcn_s_wait_asynccnt)
  __builtin_amdgcn_s_wait_asynccnt(0);
#endif
  __syncthreads();
}

/* ---------------- WMMA fragment helpers (contiguous-k layouts, wave32) ------ */

__device__ __forceinline__ int lane31() { return (int)(threadIdx.x & 31u); }

/* A fragment: 16x32 bf16 from this lane's row (contiguous K, f32 src).
   Lane layout: elements 0..15 = K [k0+16*half .. +15];
   one 64B run -> four b128 loads. */
__device__ __forceinline__ v16bf afrag_row(const float* rowp, int k0) {
  int half = lane31() >> 4;
  const float* p = rowp + k0 + 16 * half;
  float4 x0 = *(const float4*)(p);
  float4 x1 = *(const float4*)(p + 4);
  float4 y0 = *(const float4*)(p + 8);
  float4 y1 = *(const float4*)(p + 12);
  v16bf a;
  a[0] = (__bf16)x0.x; a[1] = (__bf16)x0.y; a[2]  = (__bf16)x0.z; a[3]  = (__bf16)x0.w;
  a[4] = (__bf16)x1.x; a[5] = (__bf16)x1.y; a[6]  = (__bf16)x1.z; a[7]  = (__bf16)x1.w;
  a[8] = (__bf16)y0.x; a[9] = (__bf16)y0.y; a[10] = (__bf16)y0.z; a[11] = (__bf16)y0.w;
  a[12] = (__bf16)y1.x; a[13] = (__bf16)y1.y; a[14] = (__bf16)y1.z; a[15] = (__bf16)y1.w;
  return a;
}

/* A fragment where element k lives at rowp[k * stride]; zero row if !valid. */
__device__ __forceinline__ v16bf afrag_strided(const float* rowp, int k0, int stride, bool valid) {
  int half = lane31() >> 4;
  v16bf a;
#pragma unroll
  for (int v = 0; v < 8; ++v) {
    int kb = k0 + 16 * half + 2 * v;
    float x0 = valid ? rowp[(size_t)kb * stride]       : 0.0f;
    float x1 = valid ? rowp[(size_t)(kb + 1) * stride] : 0.0f;
    a[2 * v]     = (__bf16)x0;
    a[2 * v + 1] = (__bf16)x1;
  }
  return a;
}

/* B fragment: 32x16 bf16 from row-major f32 B[K x ldb], this lane's column. */
__device__ __forceinline__ v16bf bfrag_col(const float* B, int ldb, int k0, int col) {
  int half = lane31() >> 4;
  v16bf b;
#pragma unroll
  for (int v = 0; v < 8; ++v) {
    int k = k0 + 2 * v + 16 * half;
    b[2 * v]     = (__bf16)B[(size_t)k * ldb + col];
    b[2 * v + 1] = (__bf16)B[(size_t)(k + 1) * ldb + col];
  }
  return b;
}

/* Same with column guard (columns >= ncols read as zero). */
__device__ __forceinline__ v16bf bfrag_col_guard(const float* B, int ldb, int ncols, int k0, int col) {
  int half = lane31() >> 4;
  bool ok  = col < ncols;
  v16bf b;
#pragma unroll
  for (int v = 0; v < 8; ++v) {
    int k = k0 + 2 * v + 16 * half;
    float x0 = ok ? B[(size_t)k * ldb + col]       : 0.0f;
    float x1 = ok ? B[(size_t)(k + 1) * ldb + col] : 0.0f;
    b[2 * v]     = (__bf16)x0;
    b[2 * v + 1] = (__bf16)x1;
  }
  return b;
}

__device__ __forceinline__ v8f wmma_bf16(v16bf a, v16bf b, v8f c) {
  return __builtin_amdgcn_wmma_f32_16x16x32_bf16(false, a, false, b, (short)0, c, false, false);
}

/* ---------------- K1/K6: generic f32 GEMM + bias via bf16 WMMA ----------- */
__global__ void k_gemm_bias(const float* __restrict__ A, const float* __restrict__ B,
                            const float* __restrict__ bias, float* __restrict__ C,
                            int M, int K, int Nn) {
  int wave = (int)((blockIdx.x * blockDim.x + threadIdx.x) >> 5);
  int ntn  = Nn >> 4;
  int tot  = (M >> 4) * ntn;
  if (wave >= tot) return;
  int tm = wave / ntn, tn = wave % ntn;
  int lrow = lane31() & 15, half = lane31() >> 4;
  const float* arow = A + (size_t)(tm * 16 + lrow) * K;
  int col = tn * 16 + (lane31() & 15);
  v8f acc = {};
  for (int k0 = 0; k0 < K; k0 += 32) {
    v16bf a = afrag_row(arow, k0);
    v16bf b = bfrag_col(B, Nn, k0, col);
    acc = wmma_bf16(a, b, acc);
  }
  float bv = bias ? bias[col] : 0.0f;
#pragma unroll
  for (int r = 0; r < 8; ++r) {
    int row = tm * 16 + r + 8 * half;
    C[(size_t)row * Nn + col] = acc[r] + bv;
  }
}

/* ---------------- K2a: pack U/V^T + rank-1 terms for the logits bilinear - */
__global__ void k_prep_uv(const float* __restrict__ pq_l, const float* __restrict__ pk_l,
                          const float* __restrict__ sq,   const float* __restrict__ sk,
                          const float* __restrict__ rot,  const float* __restrict__ tran,
                          const float* __restrict__ w_attn, const float* __restrict__ tpw,
                          float* __restrict__ U, float* __restrict__ Vt,
                          float* __restrict__ rowq, float* __restrict__ colk) {
  int idx = (int)(blockIdx.x * blockDim.x + threadIdx.x);
  if (idx >= kN * kH) return;
  int i = idx / kH, h = idx % kH;
  float w0 = w_attn[0], w1 = w_attn[1];
  float sp = log1pf(__expf(tpw[h]));
  float ch = 0.5f * sp * kPW * w1;
  const float* R = rot + (size_t)i * 9;
  const float* t = tran + (size_t)i * 3;
  const float* ql = pq_l + (size_t)i * (kH * kPQK * 3) + h * (kPQK * 3);
  const float* kl = pk_l + (size_t)i * (kH * kPQK * 3) + h * (kPQK * 3);
  float qg[kPQK * 3], kg[kPQK * 3];
  float q2 = 0.0f, k2 = 0.0f;
#pragma unroll
  for (int p = 0; p < kPQK; ++p)
#pragma unroll
    for (int x = 0; x < 3; ++x) {
      float gq = R[x * 3] * ql[p * 3] + R[x * 3 + 1] * ql[p * 3 + 1] + R[x * 3 + 2] * ql[p * 3 + 2] + t[x];
      float gk = R[x * 3] * kl[p * 3] + R[x * 3 + 1] * kl[p * 3 + 1] + R[x * 3 + 2] * kl[p * 3 + 2] + t[x];
      qg[p * 3 + x] = gq; kg[p * 3 + x] = gk;
      q2 += gq * gq; k2 += gk * gk;
    }
  float* u = U + ((size_t)h * kN + i) * 32;
#pragma unroll
  for (int d = 0; d < 16; ++d) u[d] = w0 * kSW * sq[((size_t)i * kH + h) * kDQK + d];
#pragma unroll
  for (int d = 0; d < 12; ++d) u[16 + d] = 2.0f * ch * qg[d];
  u[28] = u[29] = u[30] = u[31] = 0.0f;
  float* vt = Vt + (size_t)h * 32 * kN;
#pragma unroll
  for (int d = 0; d < 16; ++d) vt[(size_t)d * kN + i] = sk[((size_t)i * kH + h) * kDQK + d];
#pragma unroll
  for (int d = 0; d < 12; ++d) vt[(size_t)(16 + d) * kN + i] = kg[d];
#pragma unroll
  for (int d = 28; d < 32; ++d) vt[(size_t)d * kN + i] = 0.0f;
  rowq[(size_t)h * kN + i] = -ch * q2;
  colk[(size_t)h * kN + i] = -ch * k2;
}

/* ---------------- K2b: value points to global frame --------------------- */
__global__ void k_pv_global(const float* __restrict__ pv_l, const float* __restrict__ rot,
                            const float* __restrict__ tran, float* __restrict__ pvg) {
  int idx = (int)(blockIdx.x * blockDim.x + threadIdx.x);
  if (idx >= kN * kH * kPV) return;
  int p = idx % kPV, h = (idx / kPV) % kH, i = idx / (kPV * kH);
  const float* R = rot + (size_t)i * 9;
  const float* t = tran + (size_t)i * 3;
  const float* vl = pv_l + (size_t)i * (kH * kPV * 3) + (h * kPV + p) * 3;
  float* out = pvg + (((size_t)i * kH + h) * kPV + p) * 3;
#pragma unroll
  for (int x = 0; x < 3; ++x)
    out[x] = R[x * 3] * vl[0] + R[x * 3 + 1] * vl[1] + R[x * 3 + 2] * vl[2] + t[x];
}

/* ---------------- K3a: pair bias via async-staged z tiles ---------------- */
__global__ void k_sb_logits(const float* __restrict__ z, const float* __restrict__ Wsb,
                            const float* __restrict__ bsb, const float* __restrict__ w_attn,
                            const float* __restrict__ b_attn, float* __restrict__ logits) {
  __shared__ float tile[128 * kCZ];
  int i   = (int)blockIdx.x;
  int tid = (int)threadIdx.x;
  int wv  = tid >> 5;
  int lrow = lane31() & 15, half = lane31() >> 4, lcol = lane31() & 15;
  float w2 = w_attn[2] * kZW, ba = b_attn[0];
  float bb = (lcol < kH) ? bsb[lcol] : 0.0f;
  for (int jc = 0; jc < 4; ++jc) {
    if (jc) __syncthreads();               /* protect LDS before restaging */
    const float* src = z + ((size_t)i * kN + jc * 128) * kCZ;
#pragma unroll
    for (int it = 0; it < 16; ++it) {
      int idx4 = tid + 256 * it;           /* 4096 float4s = 128x128 f32 */
      async_copy_f4(src + (size_t)idx4 * 4, tile + (size_t)idx4 * 4);
    }
    async_join();
    const float* arow = tile + (wv * 16 + lrow) * kCZ;
    v8f acc = {};
#pragma unroll
    for (int k0 = 0; k0 < kCZ; k0 += 32) {
      v16bf a = afrag_row(arow, k0);
      v16bf b = bfrag_col_guard(Wsb, kH, kH, k0, lcol);
      acc = wmma_bf16(a, b, acc);
    }
    if (lcol < kH) {
#pragma unroll
      for (int r = 0; r < 8; ++r) {
        int j = jc * 128 + wv * 16 + r + 8 * half;
        logits[((size_t)i * kN + j) * kH + lcol] = w2 * (acc[r] + bb) + ba;
      }
    }
  }
}

/* ---------------- K3b: bilinear (scalar-QK + point cross term) ---------- */
__global__ void k_bilinear(const float* __restrict__ U, const float* __restrict__ Vt,
                           const float* __restrict__ rowq, const float* __restrict__ colk,
                           float* __restrict__ logits) {
  int wave = (int)((blockIdx.x * blockDim.x + threadIdx.x) >> 5);
  if (wave >= kH * 32 * 32) return;
  int h = wave >> 10, rem = wave & 1023, it = rem >> 5, jt = rem & 31;
  int lrow = lane31() & 15, half = lane31() >> 4, lcol = lane31() & 15;
  const float* urow = U + ((size_t)h * kN + it * 16 + lrow) * 32;
  int j = jt * 16 + lcol;
  v16bf a = afrag_row(urow, 0);
  v16bf b = bfrag_col(Vt + (size_t)h * 32 * kN, kN, 0, j);
  v8f acc = {};
  acc = wmma_bf16(a, b, acc);
  float ck = colk[(size_t)h * kN + j];
#pragma unroll
  for (int r = 0; r < 8; ++r) {
    int i = it * 16 + r + 8 * half;
    size_t off = ((size_t)i * kN + j) * kH + h;
    logits[off] += acc[r] + rowq[(size_t)h * kN + i] + ck;
  }
}

/* ---------------- K4: masked softmax over j (one wave per (i,h) row) ----- */
__global__ void k_softmax(const float* __restrict__ res_mask, float* __restrict__ logits) {
  int wave = (int)((blockIdx.x * blockDim.x + threadIdx.x) >> 5);
  if (wave >= kN * kH) return;
  int i = wave / kH, h = wave % kH;
  int lane = lane31();
  float mi = res_mask[i];
  float vals[16];
  float mx = kNEG;
#pragma unroll
  for (int t = 0; t < 16; ++t) {
    int j = t * 32 + lane;
    float l = logits[((size_t)i * kN + j) * kH + h];
    if (mi * res_mask[j] == 0.0f) l = kNEG;
    vals[t] = l;
    mx = fmaxf(mx, l);
  }
#pragma unroll
  for (int o = 16; o; o >>= 1) mx = fmaxf(mx, __shfl_xor(mx, o, 32));
  float sum = 0.0f;
#pragma unroll
  for (int t = 0; t < 16; ++t) { vals[t] = __expf(vals[t] - mx); sum += vals[t]; }
#pragma unroll
  for (int o = 16; o; o >>= 1) sum += __shfl_xor(sum, o, 32);
  float inv = 1.0f / sum;
#pragma unroll
  for (int t = 0; t < 16; ++t) {
    int j = t * 32 + lane;
    logits[((size_t)i * kN + j) * kH + h] = vals[t] * inv;
  }
}

/* ---------------- K5a: attn @ [sv | pvg]  (per head, K=512 over j) ------- */
__device__ __forceinline__ float svpv_elem(const float* sv, const float* pvg, int j, int h, int n) {
  if (n < 16) return sv[((size_t)j * kH + h) * kDV + n];
  if (n < 40) return pvg[((size_t)j * kH + h) * (kPV * 3) + (n - 16)];
  return 0.0f;
}

__global__ void k_attn_svpv(const float* __restrict__ attn, const float* __restrict__ sv,
                            const float* __restrict__ pvg, float* __restrict__ cat,
                            float* __restrict__ rptg) {
  int wave = (int)((blockIdx.x * blockDim.x + threadIdx.x) >> 5);
  if (wave >= kH * 32 * 3) return;
  int h = wave / 96, rem = wave % 96, it = rem / 3, nt = rem % 3;
  int lrow = lane31() & 15, half = lane31() >> 4, lcol = lane31() & 15;
  int i_lane = it * 16 + lrow;
  const float* abase = attn + (size_t)i_lane * kN * kH + h; /* element k at stride kH */
  int n = nt * 16 + lcol;
  v8f acc = {};
  for (int k0 = 0; k0 < kN; k0 += 32) {
    v16bf a = afrag_strided(abase, k0, kH, true);
    int hh = lane31() >> 4;
    v16bf b;
#pragma unroll
    for (int v = 0; v < 8; ++v) {
      int k = k0 + 2 * v + 16 * hh;
      b[2 * v]     = (__bf16)svpv_elem(sv, pvg, k, h, n);
      b[2 * v + 1] = (__bf16)svpv_elem(sv, pvg, k + 1, h, n);
    }
    acc = wmma_bf16(a, b, acc);
  }
#pragma unroll
  for (int r = 0; r < 8; ++r) {
    int i = it * 16 + r + 8 * half;
    if (n < 16)      cat[(size_t)i * kCAT + h * 176 + n] = acc[r];
    else if (n < 40) rptg[((size_t)i * kH + h) * 24 + (n - 16)] = acc[r];
  }
}

/* ---------------- K5b: rotate points back, norms ------------------------- */
__global__ void k_pt_final(const float* __restrict__ rptg, const float* __restrict__ rot,
                           const float* __restrict__ tran, float* __restrict__ cat) {
  int idx = (int)(blockIdx.x * blockDim.x + threadIdx.x);
  if (idx >= kN * kH * kPV) return;
  int p = idx % kPV, h = (idx / kPV) % kH, i = idx / (kPV * kH);
  const float* R = rot + (size_t)i * 9;
  const float* t = tran + (size_t)i * 3;
  const float* g = rptg + ((size_t)i * kH + h) * 24 + p * 3;
  float d0 = g[0] - t[0], d1 = g[1] - t[1], d2 = g[2] - t[2];
  float n2 = kEPS;
  float* base = cat + (size_t)i * kCAT + h * 176;
#pragma unroll
  for (int x = 0; x < 3; ++x) {
    float loc = R[x] * d0 + R[3 + x] * d1 + R[6 + x] * d2; /* R^T (g - t) */
    base[16 + p * 3 + x] = loc;
    n2 += loc * loc;
  }
  base[40 + p] = sqrtf(n2);
}

/* ---------------- K5c: r_pair = attn^T(i) @ z(i), async z staging -------- */
__global__ void k_rpair(const float* __restrict__ attn, const float* __restrict__ z,
                        float* __restrict__ cat) {
  __shared__ float tile[128 * kCZ];
  int i   = (int)blockIdx.x;
  int tid = (int)threadIdx.x;
  int wv  = tid >> 5;
  int lrow = lane31() & 15, half = lane31() >> 4, lcol = lane31() & 15;
  const float* abase = attn + (size_t)i * kN * kH + lrow; /* A[m=h][k=j], stride kH */
  int c = wv * 16 + lcol;
  v8f acc = {};
  for (int jc = 0; jc < 4; ++jc) {
    if (jc) __syncthreads();
    const float* src = z + ((size_t)i * kN + jc * 128) * kCZ;
#pragma unroll
    for (int it = 0; it < 16; ++it) {
      int idx4 = tid + 256 * it;
      async_copy_f4(src + (size_t)idx4 * 4, tile + (size_t)idx4 * 4);
    }
    async_join();
#pragma unroll
    for (int kk = 0; kk < 128; kk += 32) {
      v16bf a = afrag_strided(abase, jc * 128 + kk, kH, lrow < kH);
      v16bf b = bfrag_col(tile, kCZ, kk, c);
      acc = wmma_bf16(a, b, acc);
    }
  }
#pragma unroll
  for (int r = 0; r < 8; ++r) {
    int h = r + 8 * half;
    if (h < kH) cat[(size_t)i * kCAT + h * 176 + 48 + c] = acc[r];
  }
}

/* ---------------- host launch -------------------------------------------- */
extern "C" void kernel_launch(void* const* d_in, const int* in_sizes, int n_in,
                              void* d_out, int out_size, void* d_ws, size_t ws_size,
                              hipStream_t stream) {
  (void)in_sizes; (void)n_in; (void)out_size; (void)ws_size;
  const float* s        = (const float*)d_in[0];
  const float* z        = (const float*)d_in[1];
  const float* rot      = (const float*)d_in[2];
  const float* tran     = (const float*)d_in[3];
  const float* res_mask = (const float*)d_in[4];
  const float* Wsq = (const float*)d_in[5],  *bsq = (const float*)d_in[6];
  const float* Wsk = (const float*)d_in[7],  *bsk = (const float*)d_in[8];
  const float* Wsv = (const float*)d_in[9],  *bsv = (const float*)d_in[10];
  const float* Wsb = (const float*)d_in[11], *bsb = (const float*)d_in[12];
  const float* Wpq = (const float*)d_in[13], *bpq = (const float*)d_in[14];
  const float* Wpk = (const float*)d_in[15], *bpk = (const float*)d_in[16];
  const float* Wpv = (const float*)d_in[17], *bpv = (const float*)d_in[18];
  const float* w_attn = (const float*)d_in[19];
  const float* b_attn = (const float*)d_in[20];
  const float* tpw    = (const float*)d_in[21];
  const float* Wout = (const float*)d_in[22], *bout = (const float*)d_in[23];
  float* out = (float*)d_out;

  float* ws = (float*)d_ws;
  float* sq    = ws;               /*  98304 */
  float* sk    = sq    +  98304;   /*  98304 */
  float* sv    = sk    +  98304;   /*  98304 */
  float* pq_l  = sv    +  98304;   /*  73728 */
  float* pk_l  = pq_l  +  73728;   /*  73728 */
  float* pv_l  = pk_l  +  73728;   /* 147456 */
  float* pvg   = pv_l  + 147456;   /* 147456 */
  float* U     = pvg   + 147456;   /* 196608 */
  float* Vt    = U     + 196608;   /* 196608 */
  float* rowq  = Vt    + 196608;   /*   6144 */
  float* colk  = rowq  +   6144;   /*   6144 */
  float* rptg  = colk  +   6144;   /* 147456 */
  float* cat   = rptg  + 147456;   /* 1081344 */
  float* logits= cat   + 1081344;  /* 3145728 */

  const int TB = 256; /* 8 waves */
  auto wblocks = [](int waves) { return (waves + 7) / 8; };

  /* K1: six input projections (s @ W + b) */
  k_gemm_bias<<<wblocks(32 * 12), TB, 0, stream>>>(s, Wsq, bsq, sq,   kN, kCS, 192);
  k_gemm_bias<<<wblocks(32 * 12), TB, 0, stream>>>(s, Wsk, bsk, sk,   kN, kCS, 192);
  k_gemm_bias<<<wblocks(32 * 12), TB, 0, stream>>>(s, Wsv, bsv, sv,   kN, kCS, 192);
  k_gemm_bias<<<wblocks(32 *  9), TB, 0, stream>>>(s, Wpq, bpq, pq_l, kN, kCS, 144);
  k_gemm_bias<<<wblocks(32 *  9), TB, 0, stream>>>(s, Wpk, bpk, pk_l, kN, kCS, 144);
  k_gemm_bias<<<wblocks(32 * 18), TB, 0, stream>>>(s, Wpv, bpv, pv_l, kN, kCS, 288);

  /* K2: frame transforms + bilinear packing */
  k_pv_global<<<(kN * kH * kPV + TB - 1) / TB, TB, 0, stream>>>(pv_l, rot, tran, pvg);
  k_prep_uv<<<(kN * kH + TB - 1) / TB, TB, 0, stream>>>(pq_l, pk_l, sq, sk, rot, tran,
                                                        w_attn, tpw, U, Vt, rowq, colk);

  /* K3: logits = pair-bias + fused (scalar QK + point) bilinear + rank-1 */
  k_sb_logits<<<kN, TB, 0, stream>>>(z, Wsb, bsb, w_attn, b_attn, logits);
  k_bilinear<<<wblocks(kH * 32 * 32), TB, 0, stream>>>(U, Vt, rowq, colk, logits);

  /* K4: softmax over j */
  k_softmax<<<wblocks(kN * kH), TB, 0, stream>>>(res_mask, logits);

  /* K5: attention-weighted aggregations */
  k_attn_svpv<<<wblocks(kH * 32 * 3), TB, 0, stream>>>(logits, sv, pvg, cat, rptg);
  k_pt_final<<<(kN * kH * kPV + TB - 1) / TB, TB, 0, stream>>>(rptg, rot, tran, cat);
  k_rpair<<<kN, TB, 0, stream>>>(logits, z, cat);

  /* K6: output projection */
  k_gemm_bias<<<wblocks(32 * 24), TB, 0, stream>>>(cat, Wout, bout, out, kN, kCAT, kCS);
}
